// gatBlock_24730421690786
// MI455X (gfx1250) — compile-verified, matrix operand order checked by reading, not run on hardware
//
#include <hip/hip_runtime.h>
#include <hip/hip_fp16.h>

// ---------------------------------------------------------------------------
// GATConv (H=4, C=32, concat) + LayerNorm + ReLU for MI455X (gfx1250).
// Pipeline:
//   1) wbconv : pre-swizzle W (f32->f16) into WMMA B-fragment layout (64 KB)
//   2) init   : zero accumulators, set segment-max to enc(-inf)
//   3) gemm   : x @ W via v_wmma_f32_16x16x32_f16 (LDS-staged A tile)
//   4) scores : a_src / a_dst dot products
//   5) edgemax: segment max of leaky-relu(alpha) via encoded atomicMax
//   6) edgexp : ex = exp(alpha - max), segment sum via atomicAdd
//   7) scatter: out[dst] += x_lin[src] * (ex / denom)   (float atomics, L2)
//   8) ln     : bias + LayerNorm + ReLU, one wave32 per node
// ---------------------------------------------------------------------------

typedef __attribute__((ext_vector_type(16))) _Float16 v16h;
typedef __attribute__((ext_vector_type(8)))  _Float16 v8h;
typedef __attribute__((ext_vector_type(8)))  float    v8f;

constexpr int   kN   = 50000;
constexpr int   kE   = 800000;
constexpr int   kIN  = 256;
constexpr int   kH   = 4;
constexpr int   kC   = 32;
constexpr int   kHC  = 128;          // H*C
constexpr float kNegSlope = 0.2f;

// workspace byte offsets (all multiples of 256)
constexpr size_t OFF_WB    = 0;                         //  64 KB  f16 swizzled W
constexpr size_t OFF_ASRC  = 65536;                     // 800 KB  a_src [N,H]
constexpr size_t OFF_ADST  = 865536;                    // 800 KB  a_dst [N,H]
constexpr size_t OFF_AMAX  = 1665536;                   // 800 KB  encoded seg-max
constexpr size_t OFF_DEN   = 2465536;                   // 800 KB  softmax denom
constexpr size_t OFF_XLIN  = 3265536;                   //  25.6MB x_lin [N,128]
constexpr size_t OFF_EX    = 28865536;                  //  12.8MB ex [E,H]

// monotone float<->uint encoding so atomicMax(u32) orders floats correctly
__device__ __forceinline__ unsigned encF(float f) {
    unsigned u = __float_as_uint(f);
    return (u & 0x80000000u) ? ~u : (u | 0x80000000u);
}
__device__ __forceinline__ float decF(unsigned u) {
    return (u & 0x80000000u) ? __uint_as_float(u & 0x7FFFFFFFu)
                             : __uint_as_float(~u);
}
// encF(-inf) == 0x007FFFFF
constexpr unsigned ENC_NEG_INF = 0x007FFFFFu;

// ---- 1) pre-swizzle W into WMMA B-fragment layout ---------------------------
// flat idx = ((ct*8 + ki)*32 + lane)*16 + e,  value = W[K*128 + col]
//   K   = ki*32 + (e/8)*16 + (lane/16)*8 + (e%8)
//   col = ct*16 + (lane%16)
__global__ void gat_wbconv(const float* __restrict__ W, _Float16* __restrict__ WB) {
    int idx = blockIdx.x * blockDim.x + threadIdx.x;      // 0 .. 32767
    int e  = idx & 15;
    int L  = (idx >> 4) & 31;
    int ki = (idx >> 9) & 7;
    int ct = idx >> 12;
    int K   = ki * 32 + (e >> 3) * 16 + (L >> 4) * 8 + (e & 7);
    int col = ct * 16 + (L & 15);
    WB[idx] = (_Float16)W[K * kHC + col];
}

// ---- 2) init accumulators ---------------------------------------------------
__global__ void gat_init(float* __restrict__ out, float* __restrict__ denom,
                         unsigned* __restrict__ amaxEnc) {
    int i = blockIdx.x * 256 + threadIdx.x;
    if (i < kN * kHC) {
        out[i] = 0.0f;
    } else if (i < kN * kHC + kN * kH) {
        denom[i - kN * kHC] = 0.0f;
    } else if (i < kN * kHC + 2 * kN * kH) {
        amaxEnc[i - kN * kHC - kN * kH] = ENC_NEG_INF;
    }
}

// ---- 3) x @ W via WMMA ------------------------------------------------------
// block = 256 threads (8 waves). blockIdx.x = 16-row tile. Wave w owns cols
// [w*16, w*16+16). K loop: 8 x (16x16x32) WMMAs, f32 accumulate.
__global__ void __launch_bounds__(256) gat_gemm_wmma(
        const float* __restrict__ x, const _Float16* __restrict__ WB,
        float* __restrict__ xlin) {
    __shared__ __align__(16) _Float16 At[16 * kIN];       // 8 KB A tile (f16)

    const int tid     = threadIdx.x;
    const int rowbase = blockIdx.x * 16;

    // cooperative load + f32->f16 convert: 4096 halves, 16 per thread
    {
        const int e0 = tid * 16;
        const int r  = e0 >> 8;          // /256
        const int c  = e0 & 255;
        const float* src = x + (size_t)(rowbase + r) * kIN + c;
        #pragma unroll
        for (int i = 0; i < 16; ++i) At[e0 + i] = (_Float16)src[i];
    }
    __syncthreads();

    const int wave = tid >> 5;
    const int lane = tid & 31;
    const int m    = lane & 15;          // row (A) / col (B,D) within tile
    const int h8   = (lane >> 4) * 8;    // lane-half K offset

    const v16h* wb = (const v16h*)WB + (size_t)wave * (8 * 32) + lane;

    v8f acc = {};
    #pragma unroll
    for (int ki = 0; ki < 8; ++ki) {
        const int kb = ki * 32;
        union { v16h v; v8h p[2]; } a;
        const _Float16* row = &At[m * kIN + kb + h8];
        a.p[0] = *(const v8h*)(row);          // K = kb + h8 + 0..7
        a.p[1] = *(const v8h*)(row + 16);     // K = kb + 16 + h8 + 0..7
        v16h b = wb[ki * 32];
        acc = __builtin_amdgcn_wmma_f32_16x16x32_f16(
                  false, a.v, false, b, (short)0, acc, false, false);
    }

    // store D: lane holds col = colbase + m, rows v + (lane/16)*8
    const int colbase = wave * 16;
    float* drow = xlin + (size_t)(rowbase + (lane >> 4) * 8) * kHC + colbase + m;
    #pragma unroll
    for (int v = 0; v < 8; ++v) drow[(size_t)v * kHC] = acc[v];
}

// ---- 4) attention scores ----------------------------------------------------
__global__ void gat_scores(const float* __restrict__ xlin,
                           const float* __restrict__ att_src,
                           const float* __restrict__ att_dst,
                           float* __restrict__ a_src, float* __restrict__ a_dst) {
    int i = blockIdx.x * blockDim.x + threadIdx.x;
    if (i >= kN * kH) return;
    int n = i >> 2, h = i & 3;
    const float* r  = xlin + (size_t)n * kHC + h * kC;
    const float* as = att_src + h * kC;
    const float* ad = att_dst + h * kC;
    float s = 0.0f, d = 0.0f;
    #pragma unroll
    for (int c = 0; c < kC; ++c) { float v = r[c]; s += v * as[c]; d += v * ad[c]; }
    a_src[i] = s;
    a_dst[i] = d;
}

__device__ __forceinline__ float leaky(float a) {
    return a >= 0.0f ? a : kNegSlope * a;
}

// ---- 5) segment max ---------------------------------------------------------
__global__ void gat_edge_max(const int* __restrict__ ei,
                             const float* __restrict__ a_src,
                             const float* __restrict__ a_dst,
                             unsigned* __restrict__ amaxEnc) {
    int i = blockIdx.x * 256 + threadIdx.x;    // < E*H
    int e = i >> 2, h = i & 3;
    int s = ei[e], d = ei[kE + e];
    float al = leaky(a_src[s * kH + h] + a_dst[d * kH + h]);
    atomicMax(&amaxEnc[d * kH + h], encF(al));
}

// ---- 6) exp + segment sum ---------------------------------------------------
__global__ void gat_edge_exp(const int* __restrict__ ei,
                             const float* __restrict__ a_src,
                             const float* __restrict__ a_dst,
                             const unsigned* __restrict__ amaxEnc,
                             float* __restrict__ exbuf,
                             float* __restrict__ denom) {
    int i = blockIdx.x * 256 + threadIdx.x;    // < E*H
    int e = i >> 2, h = i & 3;
    int s = ei[e], d = ei[kE + e];
    float al = leaky(a_src[s * kH + h] + a_dst[d * kH + h]);
    float mx = decF(amaxEnc[d * kH + h]);
    float ex = __expf(al - mx);
    exbuf[i] = ex;
    atomicAdd(&denom[d * kH + h], ex);
}

// ---- 7) message scatter -----------------------------------------------------
__global__ void gat_scatter(const int* __restrict__ ei,
                            const float* __restrict__ xlin,
                            const float* __restrict__ exbuf,
                            const float* __restrict__ denom,
                            float* __restrict__ acc) {
    long long i = (long long)blockIdx.x * 256 + threadIdx.x;   // < E*128
    int e = (int)(i >> 7);
    int f = (int)(i & 127);
    int h = f >> 5;
    int s = ei[e], d = ei[kE + e];
    float coef = exbuf[(size_t)e * kH + h] / (denom[d * kH + h] + 1e-16f);
    atomicAdd(&acc[(size_t)d * kHC + f], xlin[(size_t)s * kHC + f] * coef);
}

// ---- 8) bias + LayerNorm + ReLU (one wave32 per node) -----------------------
__global__ void __launch_bounds__(256) gat_ln(float* __restrict__ out,
                                              const float* __restrict__ bias,
                                              const float* __restrict__ gamma,
                                              const float* __restrict__ beta) {
    int node = blockIdx.x * 8 + (threadIdx.x >> 5);
    int lane = threadIdx.x & 31;
    float* row = out + (size_t)node * kHC;

    float v[4];
    float s = 0.0f;
    #pragma unroll
    for (int j = 0; j < 4; ++j) {
        int f = lane + j * 32;
        v[j] = row[f] + bias[f];
        s += v[j];
    }
    #pragma unroll
    for (int o = 16; o > 0; o >>= 1) s += __shfl_xor(s, o, 32);
    float mu = s * (1.0f / 128.0f);

    float q = 0.0f;
    #pragma unroll
    for (int j = 0; j < 4; ++j) { float t = v[j] - mu; q += t * t; }
    #pragma unroll
    for (int o = 16; o > 0; o >>= 1) q += __shfl_xor(q, o, 32);
    float rs = rsqrtf(q * (1.0f / 128.0f) + 1e-5f);

    #pragma unroll
    for (int j = 0; j < 4; ++j) {
        int f = lane + j * 32;
        float y = (v[j] - mu) * rs * gamma[f] + beta[f];
        row[f] = fmaxf(y, 0.0f);
    }
}

// ---------------------------------------------------------------------------
extern "C" void kernel_launch(void* const* d_in, const int* in_sizes, int n_in,
                              void* d_out, int out_size, void* d_ws, size_t ws_size,
                              hipStream_t stream) {
    (void)in_sizes; (void)n_in; (void)out_size; (void)ws_size;
    const float* x       = (const float*)d_in[0];
    const int*   ei      = (const int*)  d_in[1];
    const float* W       = (const float*)d_in[2];
    const float* att_src = (const float*)d_in[3];
    const float* att_dst = (const float*)d_in[4];
    const float* bias    = (const float*)d_in[5];
    const float* gamma   = (const float*)d_in[6];
    const float* beta    = (const float*)d_in[7];
    float* out = (float*)d_out;

    char* ws = (char*)d_ws;
    _Float16* WB      = (_Float16*)(ws + OFF_WB);
    float*    a_src   = (float*)   (ws + OFF_ASRC);
    float*    a_dst   = (float*)   (ws + OFF_ADST);
    unsigned* amaxEnc = (unsigned*)(ws + OFF_AMAX);
    float*    denom   = (float*)   (ws + OFF_DEN);
    float*    xlin    = (float*)   (ws + OFF_XLIN);
    float*    exbuf   = (float*)   (ws + OFF_EX);

    gat_wbconv   <<<128, 256, 0, stream>>>(W, WB);
    gat_init     <<<(kN * kHC + 2 * kN * kH + 255) / 256, 256, 0, stream>>>(out, denom, amaxEnc);
    gat_gemm_wmma<<<kN / 16, 256, 0, stream>>>(x, WB, xlin);
    gat_scores   <<<(kN * kH + 255) / 256, 256, 0, stream>>>(xlin, att_src, att_dst, a_src, a_dst);
    gat_edge_max <<<(kE * kH) / 256, 256, 0, stream>>>(ei, a_src, a_dst, amaxEnc);
    gat_edge_exp <<<(kE * kH) / 256, 256, 0, stream>>>(ei, a_src, a_dst, amaxEnc, exbuf, denom);
    gat_scatter  <<<(kE * kHC) / 256, 256, 0, stream>>>(ei, xlin, exbuf, denom, out);
    gat_ln       <<<kN / 8, 256, 0, stream>>>(out, bias, gamma, beta);
}
